// AggregatedAttention_8160437862500
// MI455X (gfx1250) — matrix operand order repaired and checked by reading, not compiled, not run on hardware
//
#include <hip/hip_runtime.h>
#include <hip/hip_bf16.h>
#include <math.h>

typedef __bf16 bf16;
typedef __attribute__((ext_vector_type(16))) __bf16 v16bf;
typedef __attribute__((ext_vector_type(8)))  __bf16 v8bf;
typedef __attribute__((ext_vector_type(8)))  float  v8f;
typedef unsigned int u32x4 __attribute__((ext_vector_type(4)));
typedef int i32x4 __attribute__((ext_vector_type(4)));
typedef int i32x8 __attribute__((ext_vector_type(8)));

#define B_    8
#define Hh_   56
#define Ww_   56
#define DIM_  256
#define NH_   8
#define HD_   32
#define N_    3136
#define L_    9
#define PL_   49
#define PLP_  64               // PL padded to 4 WMMA col-tiles
#define M_    (B_ * N_)        // 25088 rows
#define MP_   392              // B_*PL_ pooled rows
#define SCALE_ 0.0625f
#define NEGBIG -1.0e30f

#if __has_include(<hip/amd_detail/amd_gfx1250_TDM.h>)
#define TDM_6ARG 1
#endif

// ---------------------------------------------------------------------------
// TDM: 1D copy of `dwords` dwords from global to LDS via tensor_load_to_lds.
// D# group0: count=1, lds_addr, global_addr, type=2.  group1: data_size=4B,
// tensor_dim0 = tile_dim0 = dwords, tensor_dim1=1, stride0 = dwords.
// Issue from one wave; caller waits s_wait_tensorcnt then block-barriers.
// ---------------------------------------------------------------------------
__device__ inline void tdm_load_1d(const void* gsrc, void* lds_dst, int dwords) {
  unsigned long long ga = (unsigned long long)(uintptr_t)gsrc;
  unsigned int glo = __builtin_amdgcn_readfirstlane((unsigned int)ga);
  unsigned int ghi = __builtin_amdgcn_readfirstlane((unsigned int)(ga >> 32));
  unsigned int la  = __builtin_amdgcn_readfirstlane((unsigned int)(uintptr_t)lds_dst);
  int nd = __builtin_amdgcn_readfirstlane(dwords);

  u32x4 g0;
  g0[0] = 1u;                                   // count=1 (valid user D#)
  g0[1] = la;                                   // lds_addr (bytes)
  g0[2] = glo;                                  // global_addr[31:0]
  g0[3] = (ghi & 0x01FFFFFFu) | (2u << 30);     // global_addr[56:32] | type=2

  i32x8 g1;
  g1[0] = (2 << 16);                            // data_size = 4 bytes
  g1[1] = (nd & 0xFFFF) << 16;                  // tensor_dim0[15:0]
  g1[2] = ((nd >> 16) & 0xFFFF) | (1 << 16);    // tensor_dim0[31:16] | tensor_dim1=1
  g1[3] = (nd & 0xFFFF) << 16;                  // tile_dim0 (16-bit)
  g1[4] = 0;                                    // tile_dim1=0, tile_dim2=0
  g1[5] = nd;                                   // tensor_dim0_stride[31:0]
  g1[6] = 0;
  g1[7] = 0;

  i32x4 zg = {};
#ifdef TDM_6ARG
  i32x8 z8 = {};
  __builtin_amdgcn_tensor_load_to_lds(g0, g1, zg, zg, z8, 0);
#else
  __builtin_amdgcn_tensor_load_to_lds(g0, g1, zg, zg, 0);
#endif
}

// ---------------------------------------------------------------------------
// WMMA K=256 tile helpers (bf16 16x16x32).
// A layout (16-bit A 16x32): lane m=lane&15; elems0..15 ->
//   K=(lane>>4)*16+e: two contiguous 16B loads.
// B layout (16-bit B 32x16): lane n=lane&15; elems0..15 -> K=(lane>>4)*16+e:
//   two contiguous 16B loads.
// ---------------------------------------------------------------------------
__device__ inline v8f wmma_tile_lds(const bf16* __restrict__ A, int m0,
                                    const bf16* bp /* LDS 16x256 panel */) {
  int lane = threadIdx.x & 31;
  const bf16* arow = A + (size_t)(m0 + (lane & 15)) * 256;
  const bf16* brow = bp + (lane & 15) * 256;
  int ka = (lane >> 4) << 4;
  int kb = (lane >> 4) << 4;
  v8f acc = {};
#pragma unroll
  for (int k0 = 0; k0 < 256; k0 += 32) {
    v16bf av, bv;
    ((v8bf*)&av)[0] = *(const v8bf*)(arow + k0 + ka);
    ((v8bf*)&av)[1] = *(const v8bf*)(arow + k0 + ka + 8);
    ((v8bf*)&bv)[0] = *(const v8bf*)(brow + k0 + kb);
    ((v8bf*)&bv)[1] = *(const v8bf*)(brow + k0 + kb + 8);
    acc = __builtin_amdgcn_wmma_f32_16x16x32_bf16(false, av, false, bv,
                                                  (short)0, acc, false, false);
  }
  return acc;
}

__device__ inline v8f wmma_tile_guarded(const bf16* __restrict__ A, int Mrows,
                                        const bf16* __restrict__ BT,
                                        int m0, int n0) {
  int lane = threadIdx.x & 31;
  int ar = m0 + (lane & 15);
  bool aok = (ar < Mrows);
  const bf16* arow = A  + (size_t)ar * 256;
  const bf16* brow = BT + (size_t)(n0 + (lane & 15)) * 256;
  int ka = (lane >> 4) << 4;
  int kb = (lane >> 4) << 4;
  v8f acc = {};
#pragma unroll
  for (int k0 = 0; k0 < 256; k0 += 32) {
    v16bf av = {};
    if (aok) {
      ((v8bf*)&av)[0] = *(const v8bf*)(arow + k0 + ka);
      ((v8bf*)&av)[1] = *(const v8bf*)(arow + k0 + ka + 8);
    }
    v16bf bv;
    ((v8bf*)&bv)[0] = *(const v8bf*)(brow + k0 + kb);
    ((v8bf*)&bv)[1] = *(const v8bf*)(brow + k0 + kb + 8);
    acc = __builtin_amdgcn_wmma_f32_16x16x32_bf16(false, av, false, bv,
                                                  (short)0, acc, false, false);
  }
  return acc;
}

// ---------------------------------------------------------------------------
// K0: transpose + bf16-convert the weights; build concatenated bias.
// WT rows: [0,256)=Wq cols, [256,768)=Wkv cols, [768,1024)=Wsr cols.
// ---------------------------------------------------------------------------
__global__ __launch_bounds__(256) void prep_w_kernel(
    const float* __restrict__ Wq,  const float* __restrict__ bq,
    const float* __restrict__ Wkv, const float* __restrict__ bkv,
    const float* __restrict__ Wsr, const float* __restrict__ bsr,
    const float* __restrict__ Wproj,
    bf16* __restrict__ WT, bf16* __restrict__ WpT, float* __restrict__ bias) {
  int row = blockIdx.x, col = threadIdx.x;
  if (row < 256) {
    WT[row * 256 + col] = (bf16)Wq[col * 256 + row];
    if (col == 0) bias[row] = bq[row];
  } else if (row < 768) {
    int c2 = row - 256;
    WT[row * 256 + col] = (bf16)Wkv[col * 512 + c2];
    if (col == 0) bias[row] = bkv[c2];
  } else if (row < 1024) {
    int c2 = row - 768;
    WT[row * 256 + col] = (bf16)Wsr[col * 256 + c2];
    if (col == 0) bias[row] = bsr[c2];
  } else {
    int c2 = row - 1024;
    WpT[c2 * 256 + col] = (bf16)Wproj[col * 256 + c2];
  }
}

__global__ __launch_bounds__(256) void cvt_x_kernel(const float* __restrict__ x,
                                                    bf16* __restrict__ xb) {
  size_t i = (size_t)blockIdx.x * 256 + threadIdx.x;
  xb[i] = (bf16)x[i];
}

__global__ __launch_bounds__(256) void zero_kp_kernel(bf16* __restrict__ kp) {
  size_t i = (size_t)blockIdx.x * 256 + threadIdx.x;
  kp[i] = (bf16)0.0f;
}

// ---------------------------------------------------------------------------
// K2: fused GEMM  x(25088x256) @ WT^T -> q / kv / gelu(sr).
// Block = 4 waves sharing one 16-col weight panel staged in LDS by the TDM.
// ---------------------------------------------------------------------------
__global__ __launch_bounds__(128) void gemm_qkvsr_kernel(
    const bf16* __restrict__ X, const bf16* __restrict__ WT,
    const float* __restrict__ bias, const float* __restrict__ qemb,
    float* __restrict__ q, float* __restrict__ kvbuf, float* __restrict__ xsr) {
  __shared__ bf16 bpanel[16 * 256];               // 8 KB weight panel
  int nt  = blockIdx.x & 63;                      // 64 N-tiles
  int mtg = blockIdx.x >> 6;                      // 392 groups of 4 M-tiles
  int wave = threadIdx.x >> 5;
  if (wave == 0) {
    tdm_load_1d(WT + (size_t)nt * 16 * 256, bpanel, 16 * 256 / 2);
    __builtin_amdgcn_s_wait_tensorcnt(0);
  }
  __syncthreads();

  int mt = mtg * 4 + wave;
  v8f acc = wmma_tile_lds(X, mt * 16, bpanel);

  int lane = threadIdx.x & 31;
  int col = nt * 16 + (lane & 15);
  int rbase = mt * 16 + ((lane >> 4) << 3);
  float bi = bias[col];
#pragma unroll
  for (int r = 0; r < 8; ++r) {
    int m = rbase + r;
    float v = acc[r] + bi;
    if (col < 256) {                              // q path (+ query embedding)
      int h = col >> 5, d = col & 31;
      int bb = m / N_, n = m - bb * N_;
      q[(((size_t)(bb * NH_ + h)) * N_ + n) * HD_ + d] = v + qemb[h * HD_ + d];
    } else if (col < 768) {                       // kv path
      kvbuf[(size_t)m * 512 + (col - 256)] = v;
    } else {                                      // sr path with exact GELU
      float g = 0.5f * v * (1.0f + erff(v * 0.70710678118654752f));
      xsr[(size_t)m * 256 + (col - 768)] = g;
    }
  }
}

// ---------------------------------------------------------------------------
// K3: 8x8 average pool + LayerNorm -> xp (f32 + bf16)
// ---------------------------------------------------------------------------
__global__ __launch_bounds__(256) void pool_ln_kernel(
    const float* __restrict__ xsr, const float* __restrict__ g,
    const float* __restrict__ be, float* __restrict__ xp, bf16* __restrict__ xpb) {
  int b = blockIdx.x / PL_, p = blockIdx.x % PL_;
  int ph = p / 7, pw = p % 7;
  int c = threadIdx.x;
  float s = 0.0f;
#pragma unroll 4
  for (int i = 0; i < 8; ++i)
    for (int j = 0; j < 8; ++j) {
      int n = (ph * 8 + i) * Ww_ + pw * 8 + j;
      s += xsr[((size_t)(b * N_ + n)) * 256 + c];
    }
  s *= (1.0f / 64.0f);
  __shared__ float r1[256], r2[256];
  r1[c] = s; r2[c] = s * s;
  __syncthreads();
  for (int st = 128; st > 0; st >>= 1) {
    if (c < st) { r1[c] += r1[c + st]; r2[c] += r2[c + st]; }
    __syncthreads();
  }
  float mu = r1[0] * (1.0f / 256.0f);
  float var = r2[0] * (1.0f / 256.0f) - mu * mu;
  float vout = (s - mu) * rsqrtf(var + 1e-5f) * g[c] + be[c];
  size_t idx = (size_t)blockIdx.x * 256 + c;
  xp[idx] = vout;
  xpb[idx] = (bf16)vout;
}

// ---------------------------------------------------------------------------
// K4: GEMM  xp(392x256) @ Wkv -> k_pool (bf16, pre-scaled, padded) + v_pool.
// ---------------------------------------------------------------------------
__global__ __launch_bounds__(128) void gemm_kvp_kernel(
    const bf16* __restrict__ XP, const bf16* __restrict__ WT,
    const float* __restrict__ bias, bf16* __restrict__ kp, float* __restrict__ vp) {
  int wid = blockIdx.x * 4 + (threadIdx.x >> 5);
  int nt = wid & 31, mt = wid >> 5;               // 32 N-tiles, 25 M-tiles
  v8f acc = wmma_tile_guarded(XP, MP_, WT + 256 * 256, mt * 16, nt * 16);
  int lane = threadIdx.x & 31;
  int col = nt * 16 + (lane & 15);                // 0..511 within kv block
  int rbase = mt * 16 + ((lane >> 4) << 3);
  float bi = bias[256 + col];
#pragma unroll
  for (int r = 0; r < 8; ++r) {
    int m = rbase + r;
    if (m >= MP_) continue;
    int b = m / PL_, p = m - b * PL_;
    float v = acc[r] + bi;
    if (col < 256) {                              // k_pool, fold in SCALE
      int h = col >> 5, d = col & 31;
      kp[(((size_t)(b * NH_ + h)) * PLP_ + p) * HD_ + d] = (bf16)(v * SCALE_);
    } else {                                      // v_pool (B,PL,NH,HD)
      int h = (col - 256) >> 5, d = col & 31;
      vp[(((size_t)(b * PL_ + p)) * NH_ + h) * HD_ + d] = v;
    }
  }
}

// ---------------------------------------------------------------------------
// K5: attention core. One wave per (b,h, 16-query tile).
//  - pooled logits via 4x wmma (q 16x32 @ k_pool^T 32x64) -> LDS
//  - local window dots / learnable tokens / joint softmax(58) / weighted sums
// ---------------------------------------------------------------------------
__global__ __launch_bounds__(32) void attn_kernel(
    const float* __restrict__ qbuf, const float* __restrict__ kv,
    const bf16* __restrict__ kp, const float* __restrict__ vpool,
    const float* __restrict__ pbp, const float* __restrict__ pbl,
    const float* __restrict__ ltok, const float* __restrict__ lbias,
    bf16* __restrict__ outb) {
  __shared__ float qs[16][32];
  __shared__ float as[16][64];
  int lane = threadIdx.x;
  int bh = blockIdx.x / 196, tile = blockIdx.x % 196;
  int b = bh >> 3, h = bh & 7;
  int n0 = tile * 16;

  {  // load q tile (16 rows x 32 dims) into LDS
    int r = lane >> 1, hf = lane & 1;
    const float* qp = qbuf + (((size_t)bh) * N_ + (n0 + r)) * HD_ + hf * 16;
    float* dst = &qs[r][hf * 16];
#pragma unroll
    for (int j = 0; j < 4; ++j) ((float4*)dst)[j] = ((const float4*)qp)[j];
  }
  __syncthreads();

  union { v16bf v; bf16 e[16]; } qa;
  {
    int r = lane & 15, kb = (lane >> 4) << 4;
#pragma unroll
    for (int e = 0; e < 8; ++e) qa.e[e] = (bf16)qs[r][kb + e];
#pragma unroll
    for (int e = 0; e < 8; ++e) qa.e[8 + e] = (bf16)qs[r][8 + kb + e];
  }

  const bf16* kpb = kp + (size_t)bh * PLP_ * HD_;
#pragma unroll
  for (int t = 0; t < 4; ++t) {
    int p = t * 16 + (lane & 15);
    const bf16* bp = kpb + (size_t)p * HD_ + ((lane >> 4) << 4);
    v16bf bv;
    ((v8bf*)&bv)[0] = *(const v8bf*)(bp);
    ((v8bf*)&bv)[1] = *(const v8bf*)(bp + 8);
    v8f acc = {};
    acc = __builtin_amdgcn_wmma_f32_16x16x32_bf16(false, qa.v, false, bv,
                                                  (short)0, acc, false, false);
    int row0 = (lane >> 4) << 3;
    if (p < PL_) {
      float pb = pbp[h * PL_ + p];
#pragma unroll
      for (int r2 = 0; r2 < 8; ++r2) as[row0 + r2][p] = acc[r2] + pb;
    }
  }
  __syncthreads();

  int r = lane >> 1, hf = lane & 1;
  int n = n0 + r;
  int y = n / Ww_, x = n - y * Ww_;
  const float* qrow = &qs[r][hf * 16];

  float sl[9], la[9];
  bool inb[9];
  int nn_[9];
#pragma unroll
  for (int l = 0; l < 9; ++l) {
    int yy = y + l / 3 - 1, xx = x + (l % 3) - 1;
    bool ok = (yy >= 0) && (yy < Hh_) && (xx >= 0) && (xx < Ww_);
    inb[l] = ok;
    int nn = yy * Ww_ + xx;
    nn_[l] = nn;
    float s = 0.0f, t = 0.0f;
    if (ok) {
      const float* kk = kv + ((size_t)(b * N_ + nn)) * 512 + h * HD_ + hf * 16;
#pragma unroll
      for (int d = 0; d < 16; ++d) s += qrow[d] * kk[d];
    }
#pragma unroll
    for (int d = 0; d < 16; ++d)
      t += qrow[d] * ltok[h * (HD_ * L_) + (hf * 16 + d) * L_ + l];
    s += __shfl_xor(s, 1, 32);
    t += __shfl_xor(t, 1, 32);
    sl[l] = ok ? (s * SCALE_ + pbl[h * L_ + l]) : NEGBIG;
    la[l] = t + lbias[h * L_ + l];
  }

  float mx = NEGBIG;
#pragma unroll
  for (int l = 0; l < 9; ++l) mx = fmaxf(mx, sl[l]);
  int p0 = hf ? 25 : 0, p1 = hf ? 49 : 25;
  for (int p = p0; p < p1; ++p) mx = fmaxf(mx, as[r][p]);
  mx = fmaxf(mx, __shfl_xor(mx, 1, 32));

  float sum = 0.0f;
  if (hf == 0) {
#pragma unroll
    for (int l = 0; l < 9; ++l) sum += expf(sl[l] - mx);
  }
  for (int p = p0; p < p1; ++p) sum += expf(as[r][p] - mx);
  sum += __shfl_xor(sum, 1, 32);
  float inv = 1.0f / sum;

  float o[16];
#pragma unroll
  for (int d = 0; d < 16; ++d) o[d] = 0.0f;

  for (int p = 0; p < PL_; ++p) {                 // pooled value aggregation
    float a = expf(as[r][p] - mx) * inv;
    const float* vv = vpool + (((size_t)(b * PL_ + p)) * NH_ + h) * HD_ + hf * 16;
#pragma unroll
    for (int d = 0; d < 16; ++d) o[d] += a * vv[d];
  }
#pragma unroll
  for (int l = 0; l < 9; ++l) {                   // local window aggregation
    if (!inb[l]) continue;
    float A = la[l] + expf(sl[l] - mx) * inv;
    const float* vv = kv + ((size_t)(b * N_ + nn_[l])) * 512 + 256 + h * HD_ + hf * 16;
#pragma unroll
    for (int d = 0; d < 16; ++d) o[d] += A * vv[d];
  }

  bf16* ob = outb + ((size_t)(b * N_ + n)) * 256 + h * HD_ + hf * 16;
#pragma unroll
  for (int d = 0; d < 16; ++d) ob[d] = (bf16)o[d];
}

// ---------------------------------------------------------------------------
// K6: final projection GEMM  out(25088x256) @ Wproj -> d_out (f32).
// Same TDM-staged weight panel scheme as K2.
// ---------------------------------------------------------------------------
__global__ __launch_bounds__(128) void gemm_proj_kernel(
    const bf16* __restrict__ OUTB, const bf16* __restrict__ WpT,
    const float* __restrict__ bproj, float* __restrict__ dout) {
  __shared__ bf16 bpanel[16 * 256];
  int nt  = blockIdx.x & 15;                      // 16 N-tiles
  int mtg = blockIdx.x >> 4;                      // 392 groups
  int wave = threadIdx.x >> 5;
  if (wave == 0) {
    tdm_load_1d(WpT + (size_t)nt * 16 * 256, bpanel, 16 * 256 / 2);
    __builtin_amdgcn_s_wait_tensorcnt(0);
  }
  __syncthreads();

  int mt = mtg * 4 + wave;
  v8f acc = wmma_tile_lds(OUTB, mt * 16, bpanel);

  int lane = threadIdx.x & 31;
  int col = nt * 16 + (lane & 15);
  int rbase = mt * 16 + ((lane >> 4) << 3);
  float bi = bproj[col];
#pragma unroll
  for (int r = 0; r < 8; ++r) {
    int m = rbase + r;
    dout[(size_t)m * 256 + col] = acc[r] + bi;
  }
}

// ---------------------------------------------------------------------------
extern "C" void kernel_launch(void* const* d_in, const int* in_sizes, int n_in,
                              void* d_out, int out_size, void* d_ws, size_t ws_size,
                              hipStream_t stream) {
  const float* x     = (const float*)d_in[0];
  const float* Wq    = (const float*)d_in[1];
  const float* bq    = (const float*)d_in[2];
  const float* Wkv   = (const float*)d_in[3];
  const float* bkv   = (const float*)d_in[4];
  const float* qemb  = (const float*)d_in[5];
  const float* Wsr   = (const float*)d_in[6];
  const float* bsr   = (const float*)d_in[7];
  const float* ln_g  = (const float*)d_in[8];
  const float* ln_b  = (const float*)d_in[9];
  const float* pbp   = (const float*)d_in[10];
  const float* pbl   = (const float*)d_in[11];
  const float* ltok  = (const float*)d_in[12];
  const float* lbias = (const float*)d_in[13];
  const float* Wproj = (const float*)d_in[14];
  const float* bproj = (const float*)d_in[15];

  char* w = (char*)d_ws;
  auto alloc = [&](size_t bytes) {
    char* p = w;
    w += (bytes + 255) & ~(size_t)255;
    return p;
  };
  bf16*  x_bf   = (bf16*)alloc((size_t)M_ * 256 * 2);        // reused as out_bf
  bf16*  WT     = (bf16*)alloc((size_t)1024 * 256 * 2);
  bf16*  WpT    = (bf16*)alloc((size_t)256 * 256 * 2);
  float* biasc  = (float*)alloc((size_t)1024 * 4);
  float* qbuf   = (float*)alloc((size_t)B_ * NH_ * N_ * HD_ * 4);
  float* kvbuf  = (float*)alloc((size_t)M_ * 512 * 4);
  float* xsr    = (float*)alloc((size_t)M_ * 256 * 4);
  float* xp     = (float*)alloc((size_t)MP_ * 256 * 4);
  bf16*  xp_bf  = (bf16*)alloc((size_t)MP_ * 256 * 2);
  bf16*  kp_bf  = (bf16*)alloc((size_t)B_ * NH_ * PLP_ * HD_ * 2);
  float* vpool  = (float*)alloc((size_t)B_ * PL_ * NH_ * HD_ * 4);
  bf16*  out_bf = x_bf;   // x_bf dead after gemm_qkvsr; safe to alias

  prep_w_kernel<<<1280, 256, 0, stream>>>(Wq, bq, Wkv, bkv, Wsr, bsr, Wproj,
                                          WT, WpT, biasc);
  cvt_x_kernel<<<M_, 256, 0, stream>>>(x, x_bf);
  zero_kp_kernel<<<512, 256, 0, stream>>>(kp_bf);
  gemm_qkvsr_kernel<<<25088, 128, 0, stream>>>(x_bf, WT, biasc, qemb,
                                               qbuf, kvbuf, xsr);
  pool_ln_kernel<<<MP_, 256, 0, stream>>>(xsr, ln_g, ln_b, xp, xp_bf);
  gemm_kvp_kernel<<<200, 128, 0, stream>>>(xp_bf, WT, biasc, kp_bf, vpool);
  attn_kernel<<<12544, 32, 0, stream>>>(qbuf, kvbuf, kp_bf, vpool,
                                        pbp, pbl, ltok, lbias, out_bf);
  gemm_proj_kernel<<<6272, 128, 0, stream>>>(out_bf, WpT, bproj, (float*)d_out);
}
